// SelfAttention_26628797235450
// MI455X (gfx1250) — matrix operand order repaired and checked by reading, not compiled, not run on hardware
//
#include <hip/hip_runtime.h>
#include <cstddef>

#define DIMC 1024
#define HC   16
#define DHC  64
#define BC   2
#define TC   2048
#define MTOT (BC*TC)          // 4096 rows
#define MTILES (MTOT/16)      // 256
#define QTILES (TC/16)        // 128
#define SCALE_F 0.125f        // 1/sqrt(64)

typedef __attribute__((ext_vector_type(16))) __bf16 v16bf;
typedef __attribute__((ext_vector_type(8)))  float  v8f;

union FragBf { v16bf v; uint4 q[2]; unsigned short s[16]; };
union AccF   { v8f v; float f[8]; };
union Pack8  { unsigned short s[8]; uint4 q; };

__device__ __forceinline__ unsigned short f32_to_bf16(float f) {
  union { float f; unsigned int u; } c; c.f = f;
  unsigned int r = c.u + 0x7FFFu + ((c.u >> 16) & 1u);
  return (unsigned short)(r >> 16);
}

__device__ __forceinline__ v8f wmma_bf16(const FragBf& a, const FragBf& b, v8f c) {
  return __builtin_amdgcn_wmma_f32_16x16x32_bf16(false, a.v, false, b.v,
                                                 (short)0, c, false, false);
}

// ---------------- Kernel 1a: f32 -> bf16 (vectorized x4) ----------------
__global__ void k_cvt_bf16(const float* __restrict__ s,
                           unsigned short* __restrict__ d, int n4) {
  int i = blockIdx.x * blockDim.x + threadIdx.x;
  if (i < n4) {
    float4 f = reinterpret_cast<const float4*>(s)[i];
    union { unsigned short s[4]; uint2 u; } p;
    p.s[0] = f32_to_bf16(f.x); p.s[1] = f32_to_bf16(f.y);
    p.s[2] = f32_to_bf16(f.z); p.s[3] = f32_to_bf16(f.w);
    reinterpret_cast<uint2*>(d)[i] = p.u;
  }
}

// ---------------- Kernel 1b: rope cos/sin tables ----------------
__global__ void k_rope_tab(const float* __restrict__ rot,
                           float* __restrict__ c, float* __restrict__ s, int n) {
  int i = blockIdx.x * blockDim.x + threadIdx.x;
  if (i < n) { float p = rot[i]; c[i] = __cosf(p); s[i] = __sinf(p); }
}

// ---------------- Kernel 2: QKV projection + fused RoPE ----------------
// One wave computes a 16(M) x 64(N=one head) tile of X @ W^T.
// Q,K stored [B,H,T,DH]; V stored transposed [B,H,DH,T] (packed b128 stores).
__global__ __launch_bounds__(128) void k_gemm_qkv(
    const unsigned short* __restrict__ xbf,
    const unsigned short* __restrict__ wq,
    const unsigned short* __restrict__ wk,
    const unsigned short* __restrict__ wv,
    const float* __restrict__ cosr, const float* __restrict__ sinr,
    unsigned short* __restrict__ qout, unsigned short* __restrict__ kout,
    unsigned short* __restrict__ vtout) {
  int wave = blockIdx.x * 4 + (threadIdx.x >> 5);
  int lane = threadIdx.x & 31;
  int mat  = wave / (MTILES * HC);
  int rem  = wave % (MTILES * HC);
  int head = rem / MTILES;
  int mtile = rem % MTILES;
  const unsigned short* W = (mat == 0) ? wq : (mat == 1) ? wk : wv;

  int m0 = mtile * 16;
  int lm = lane & 15;
  bool hi = lane >= 16;

  AccF acc[4];
#pragma unroll
  for (int nt = 0; nt < 4; ++nt)
#pragma unroll
    for (int r = 0; r < 8; ++r) acc[nt].f[r] = 0.f;

  const unsigned short* arow = xbf + (size_t)(m0 + lm) * DIMC;
  for (int k0 = 0; k0 < DIMC; k0 += 32) {
    FragBf a;
    int ka = k0 + (hi ? 8 : 0);
    a.q[0] = *reinterpret_cast<const uint4*>(arow + ka);
    a.q[1] = *reinterpret_cast<const uint4*>(arow + ka + 16);
    int kb = k0 + (hi ? 8 : 0);
#pragma unroll
    for (int nt = 0; nt < 4; ++nt) {
      FragBf b;
      const unsigned short* brow =
          W + (size_t)(head * DHC + nt * 16 + lm) * DIMC + kb;
      b.q[0] = *reinterpret_cast<const uint4*>(brow);
      b.q[1] = *reinterpret_cast<const uint4*>(brow + 16);
      acc[nt].v = wmma_bf16(a, b, acc[nt].v);
    }
  }

  // RoPE epilogue for Q,K: pair columns (d, d+32) -> tiles (0,2) and (1,3)
  if (mat < 2) {
#pragma unroll
    for (int pr = 0; pr < 2; ++pr) {
      int d = pr * 16 + lm;
#pragma unroll
      for (int r = 0; r < 8; ++r) {
        int row = m0 + r + (hi ? 8 : 0);          // flattened b*T + t
        size_t base = (size_t)row * DHC;
        float c1 = cosr[base + d],      s1 = sinr[base + d];
        float c2 = cosr[base + d + 32], s2 = sinr[base + d + 32];
        float av = acc[pr].f[r], bv = acc[pr + 2].f[r];
        acc[pr].f[r]     = av * c1 - bv * s1;
        acc[pr + 2].f[r] = bv * c2 + av * s2;
      }
    }
  }

  int b  = m0 / TC;
  int t0 = m0 % TC;
  if (mat < 2) {
    unsigned short* out = (mat == 0) ? qout : kout;
    size_t hb = ((size_t)b * HC + head) * TC;
#pragma unroll
    for (int nt = 0; nt < 4; ++nt)
#pragma unroll
      for (int r = 0; r < 8; ++r) {
        int t = t0 + r + (hi ? 8 : 0);
        out[(hb + t) * DHC + nt * 16 + lm] = f32_to_bf16(acc[nt].f[r]);
      }
  } else {
    size_t hb = ((size_t)b * HC + head) * DHC;
    int t = t0 + (hi ? 8 : 0);
#pragma unroll
    for (int nt = 0; nt < 4; ++nt) {
      int d = nt * 16 + lm;
      Pack8 pk;
#pragma unroll
      for (int r = 0; r < 8; ++r) pk.s[r] = f32_to_bf16(acc[nt].f[r]);
      *reinterpret_cast<uint4*>(vtout + (hb + d) * TC + t) = pk.q;
    }
  }
}

// ---------------- Kernel 3: windowed flash attention ----------------
__global__ __launch_bounds__(128) void k_attn(
    const unsigned short* __restrict__ qbf,
    const unsigned short* __restrict__ kbf,
    const unsigned short* __restrict__ vt,
    unsigned short* __restrict__ ctx) {
  __shared__ unsigned short Plds[4][16 * 32];
  int wid  = threadIdx.x >> 5;
  int lane = threadIdx.x & 31;
  int gwave = blockIdx.x * 4 + wid;
  int bh    = gwave / QTILES;
  int qtile = gwave % QTILES;
  int b = bh / HC, h = bh % HC;
  size_t baseQK = ((size_t)b * HC + h) * TC * DHC;
  const unsigned short* Q  = qbf + baseQK;
  const unsigned short* K  = kbf + baseQK;
  const unsigned short* Vt = vt + baseQK;   // [DH, T] per head
  int m0 = qtile * 16;
  int lm = lane & 15;
  bool hiL = lane >= 16;
  unsigned short* P = Plds[wid];

  FragBf qa[2];
  {
    const unsigned short* qrow = Q + (size_t)(m0 + lm) * DHC;
#pragma unroll
    for (int kc = 0; kc < 2; ++kc) {
      int ka = kc * 32 + (hiL ? 8 : 0);
      qa[kc].q[0] = *reinterpret_cast<const uint4*>(qrow + ka);
      qa[kc].q[1] = *reinterpret_cast<const uint4*>(qrow + ka + 16);
    }
  }

  AccF o[4];
  float rowM[8], rowL[8];
#pragma unroll
  for (int nt = 0; nt < 4; ++nt)
#pragma unroll
    for (int r = 0; r < 8; ++r) o[nt].f[r] = 0.f;
#pragma unroll
  for (int r = 0; r < 8; ++r) { rowM[r] = -1e30f; rowL[r] = 0.f; }

  for (int p = 0; p < 5; ++p) {
    int kb0 = m0 - 144 + p * 32;               // first key of this pair
    AccF s0, s1;
#pragma unroll
    for (int r = 0; r < 8; ++r) { s0.f[r] = 0.f; s1.f[r] = 0.f; }
#pragma unroll
    for (int tile = 0; tile < 2; ++tile) {
      int key  = kb0 + tile * 16 + lm;
      int keyc = key > 0 ? key : 0;
      const unsigned short* krow = K + (size_t)keyc * DHC;
      AccF* st = tile ? &s1 : &s0;
#pragma unroll
      for (int kc = 0; kc < 2; ++kc) {
        FragBf kf;
        int kk = kc * 32 + (hiL ? 8 : 0);
        kf.q[0] = *reinterpret_cast<const uint4*>(krow + kk);
        kf.q[1] = *reinterpret_cast<const uint4*>(krow + kk + 16);
        st->v = wmma_bf16(qa[kc], kf, st->v);
      }
    }

    // scale + mask + online softmax (per-row butterfly over 16-lane halves)
#pragma unroll
    for (int r = 0; r < 8; ++r) {
      int qrow = m0 + r + (hiL ? 8 : 0);
      int key0 = kb0 + lm;
      int key1 = kb0 + 16 + lm;
      float v0 = s0.f[r] * SCALE_F, v1 = s1.f[r] * SCALE_F;
      bool ok0 = (key0 >= 0) & (key0 <= qrow) & (key0 >= qrow - 128);
      bool ok1 = (key1 >= 0) & (key1 <= qrow) & (key1 >= qrow - 128);
      v0 = ok0 ? v0 : -1e30f;
      v1 = ok1 ? v1 : -1e30f;
      float mx = fmaxf(v0, v1);
#pragma unroll
      for (int off = 1; off < 16; off <<= 1)
        mx = fmaxf(mx, __shfl_xor(mx, off, 32));
      float newM = fmaxf(rowM[r], mx);
      float alpha = __expf(rowM[r] - newM);
      float p0 = ok0 ? __expf(v0 - newM) : 0.f;
      float p1 = ok1 ? __expf(v1 - newM) : 0.f;
      float sum = p0 + p1;
#pragma unroll
      for (int off = 1; off < 16; off <<= 1)
        sum += __shfl_xor(sum, off, 32);
      rowL[r] = rowL[r] * alpha + sum;
      rowM[r] = newM;
#pragma unroll
      for (int nt = 0; nt < 4; ++nt) o[nt].f[r] *= alpha;
      int m = r + (hiL ? 8 : 0);
      P[m * 32 + lm]      = f32_to_bf16(p0);
      P[m * 32 + 16 + lm] = f32_to_bf16(p1);
    }
    asm volatile("s_wait_dscnt 0" ::: "memory");   // LDS stores -> loads

    // Re-stage P as a 16x32 A-fragment from LDS
    FragBf pa;
    {
      int off = hiL ? 16 : 0;
      pa.q[0] = *reinterpret_cast<const uint4*>(P + lm * 32 + off);
      pa.q[1] = *reinterpret_cast<const uint4*>(P + lm * 32 + 8 + off);
    }

    // O += P @ V  (V transposed -> contiguous B-frag loads)
    int tv  = kb0 + (hiL ? 16 : 0);
    int tvc = tv > 0 ? tv : 0;   // masked keys have p==0, garbage V harmless
#pragma unroll
    for (int nt = 0; nt < 4; ++nt) {
      FragBf vf;
      const unsigned short* vrow = Vt + (size_t)(nt * 16 + lm) * TC + tvc;
      vf.q[0] = *reinterpret_cast<const uint4*>(vrow);
      vf.q[1] = *reinterpret_cast<const uint4*>(vrow + 8);
      o[nt].v = wmma_bf16(pa, vf, o[nt].v);
    }
  }

  // finalize: ctx[(b*T + t), h*64 + d] (bf16, row-major [M, DIM])
#pragma unroll
  for (int r = 0; r < 8; ++r) {
    float inv = 1.f / rowL[r];
    int t = m0 + r + (hiL ? 8 : 0);
    size_t row = (size_t)b * TC + t;
#pragma unroll
    for (int nt = 0; nt < 4; ++nt)
      ctx[row * DIMC + h * DHC + nt * 16 + lm] =
          f32_to_bf16(o[nt].f[r] * inv);
  }
}

// ---------------- Kernel 4: output projection -> f32 ----------------
__global__ __launch_bounds__(128) void k_gemm_out(
    const unsigned short* __restrict__ ctx,
    const unsigned short* __restrict__ wo,
    float* __restrict__ out) {
  int wave = blockIdx.x * 4 + (threadIdx.x >> 5);
  int lane = threadIdx.x & 31;
  int mtile = wave / 16;
  int n64   = wave % 16;
  int m0 = mtile * 16;
  int lm = lane & 15;
  bool hi = lane >= 16;

  AccF acc[4];
#pragma unroll
  for (int nt = 0; nt < 4; ++nt)
#pragma unroll
    for (int r = 0; r < 8; ++r) acc[nt].f[r] = 0.f;

  const unsigned short* arow = ctx + (size_t)(m0 + lm) * DIMC;
  for (int k0 = 0; k0 < DIMC; k0 += 32) {
    FragBf a;
    int ka = k0 + (hi ? 8 : 0);
    a.q[0] = *reinterpret_cast<const uint4*>(arow + ka);
    a.q[1] = *reinterpret_cast<const uint4*>(arow + ka + 16);
    int kb = k0 + (hi ? 8 : 0);
#pragma unroll
    for (int nt = 0; nt < 4; ++nt) {
      FragBf b;
      const unsigned short* brow =
          wo + (size_t)(n64 * 64 + nt * 16 + lm) * DIMC + kb;
      b.q[0] = *reinterpret_cast<const uint4*>(brow);
      b.q[1] = *reinterpret_cast<const uint4*>(brow + 16);
      acc[nt].v = wmma_bf16(a, b, acc[nt].v);
    }
  }
#pragma unroll
  for (int nt = 0; nt < 4; ++nt)
#pragma unroll
    for (int r = 0; r < 8; ++r) {
      int row = m0 + r + (hi ? 8 : 0);
      out[(size_t)row * DIMC + n64 * 64 + nt * 16 + lm] = acc[nt].f[r];
    }
}

// ---------------- host ----------------
extern "C" void kernel_launch(void* const* d_in, const int* in_sizes, int n_in,
                              void* d_out, int out_size, void* d_ws, size_t ws_size,
                              hipStream_t stream) {
  (void)in_sizes; (void)n_in; (void)out_size; (void)ws_size;
  const float* x   = (const float*)d_in[0];
  // d_in[1] = attention_mask (all ones) -> window handled analytically
  const float* rot = (const float*)d_in[2];
  const float* Wq  = (const float*)d_in[3];
  const float* Wk  = (const float*)d_in[4];
  const float* Wv  = (const float*)d_in[5];
  const float* Wo  = (const float*)d_in[6];
  float* out = (float*)d_out;

  char* ws = (char*)d_ws;
  const size_t SZ_X   = (size_t)MTOT * DIMC * 2;      // 8 MiB bf16
  const size_t SZ_W   = (size_t)DIMC * DIMC * 2;      // 2 MiB bf16
  const size_t SZ_HD  = (size_t)BC * HC * TC * DHC * 2; // 8 MiB bf16
  const size_t SZ_ROT = (size_t)BC * TC * DHC * 4;    // 1 MiB f32
  unsigned short* xbf  = (unsigned short*)(ws);
  unsigned short* wqb  = (unsigned short*)(ws + SZ_X);
  unsigned short* wkb  = (unsigned short*)(ws + SZ_X + SZ_W);
  unsigned short* wvb  = (unsigned short*)(ws + SZ_X + 2 * SZ_W);
  unsigned short* wob  = (unsigned short*)(ws + SZ_X + 3 * SZ_W);
  unsigned short* qbf  = (unsigned short*)(ws + SZ_X + 4 * SZ_W);
  unsigned short* kbf  = (unsigned short*)(ws + SZ_X + 4 * SZ_W + SZ_HD);
  unsigned short* vtbf = (unsigned short*)(ws + SZ_X + 4 * SZ_W + 2 * SZ_HD);
  unsigned short* ctx  = (unsigned short*)(ws + SZ_X + 4 * SZ_W + 3 * SZ_HD);
  float* cosr = (float*)(ws + SZ_X + 4 * SZ_W + 4 * SZ_HD);
  float* sinr = (float*)(ws + SZ_X + 4 * SZ_W + 4 * SZ_HD + SZ_ROT);

  // 1) precision conversion + rope tables
  {
    int nx4 = MTOT * DIMC / 4;
    k_cvt_bf16<<<(nx4 + 255) / 256, 256, 0, stream>>>(x, xbf, nx4);
    int nw4 = DIMC * DIMC / 4;
    k_cvt_bf16<<<(nw4 + 255) / 256, 256, 0, stream>>>(Wq, wqb, nw4);
    k_cvt_bf16<<<(nw4 + 255) / 256, 256, 0, stream>>>(Wk, wkb, nw4);
    k_cvt_bf16<<<(nw4 + 255) / 256, 256, 0, stream>>>(Wv, wvb, nw4);
    k_cvt_bf16<<<(nw4 + 255) / 256, 256, 0, stream>>>(Wo, wob, nw4);
    int nr = BC * TC * DHC;
    k_rope_tab<<<(nr + 255) / 256, 256, 0, stream>>>(rot, cosr, sinr, nr);
  }
  // 2) QKV projection (+RoPE, V transposed): 3*16*256 waves, 4 waves/block
  k_gemm_qkv<<<(3 * HC * MTILES) / 4, 128, 0, stream>>>(
      xbf, wqb, wkb, wvb, cosr, sinr, qbf, kbf, vtbf);
  // 3) windowed attention: B*H*QTILES waves
  k_attn<<<(BC * HC * QTILES) / 4, 128, 0, stream>>>(qbf, kbf, vtbf, ctx);
  // 4) output projection: 256*16 waves
  k_gemm_out<<<(MTILES * 16) / 4, 128, 0, stream>>>(ctx, wob, out);
}
